// PairformerStack_7026566497090
// MI455X (gfx1250) — matrix operand order repaired and checked by reading, not compiled, not run on hardware
//
#include <hip/hip_runtime.h>
#include <hip/hip_bf16.h>

typedef __attribute__((ext_vector_type(16))) __bf16 v16bf;
typedef __attribute__((ext_vector_type(8)))  float  v8f;

#define NTOK  160
#define DP_   128
#define DS_   384
#define NHEAD 8
#define NN_   (NTOK * NTOK)      // 25600
#define PAIR_ELEMS ((size_t)NN_ * DP_)   // 3,276,800
#define SING_ELEMS ((size_t)NTOK * DS_)  // 61,440

// ------------------------------ device helpers ------------------------------

__device__ inline v8f zero8() {
  v8f z;
  #pragma unroll
  for (int e = 0; e < 8; ++e) z[e] = 0.0f;
  return z;
}

__device__ inline float sigm_f(float x) { return 1.0f / (1.0f + __expf(-x)); }
__device__ inline float silu_f(float x) { return x * sigm_f(x); }

__device__ inline v8f wmma_bf16(v16bf a, v16bf b, v8f c) {
  return __builtin_amdgcn_wmma_f32_16x16x32_bf16(
      /*neg_a=*/false, a, /*neg_b=*/false, b,
      /*c_mod=*/(short)0, c, /*reuse_a=*/false, /*reuse_b=*/false);
}

// A-fragment (16x32, bf16): lane r = l&15, half h = l>>4.
// elems[0..7]  = row[8h .. 8h+7], elems[8..15] = row[16+8h .. 23+8h].
__device__ inline v16bf load_frag_A(const __bf16* p, int h) {
  union { v16bf v; uint4 u[2]; } t;
  t.u[0] = *(const uint4*)(p + 8 * h);
  t.u[1] = *(const uint4*)(p + 16 + 8 * h);
  return t.v;
}

// B-fragment (32x16, bf16) from Bt[n][k] layout: lane col c = l&15, half h.
// elems[0..7] = Bt[c][8h .. 8h+7], elems[8..15] = Bt[c][16+8h .. 23+8h].
__device__ inline v16bf load_frag_B(const __bf16* p, int h) {
  union { v16bf v; uint4 u[2]; } t;
  t.u[0] = *(const uint4*)(p + 8 * h);
  t.u[1] = *(const uint4*)(p + 16 + 8 * h);
  return t.v;
}

// ------------------------------ GEMM kernel ---------------------------------
// C[M,Nc] = A_bf16[M,K] @ B_f32[K,Nc]  (weights converted to bf16 on stage)
// 128 threads = 4 waves. Block tile 64x64, wave tile 16x64 (4 accumulators).
// A staged with 16B uint4 loads; B staged with 16B float4 loads + packed cvt.

enum {
  EP_F32 = 0,          // outF = acc
  EP_BF16,             // outB = acc
  EP_SILU_BF16,        // outB = silu(acc)
  EP_SIG_F32,          // outF = sigmoid(acc)
  EP_BIAS_SILU_BF16,   // outB = silu(acc + bias[col])
  EP_BIAS_ADD,         // resid += acc + bias[col]
  EP_ADD,              // resid += acc
  EP_GATE_ADD,         // resid += acc * gate
  EP_GATE_ADD_T        // resid[transposed row] += acc * gate
};

__global__ __launch_bounds__(128) void gemm_bf16_k(
    const __bf16* __restrict__ A, const float* __restrict__ B,
    int M, int K, int Nc,
    float* outF, __bf16* outB, const float* bias, const float* gate,
    float* resid, int mode, int transDim)
{
  __shared__ __align__(16) __bf16 As[64 * 32];   // [row][k]
  __shared__ __align__(16) __bf16 Bs[64 * 32];   // Bt: [col][k]
  const int tid  = threadIdx.x;
  const int lane = tid & 31, wave = tid >> 5;
  const int r = lane & 15, hh = lane >> 4;
  const int m0 = blockIdx.y * 64, n0 = blockIdx.x * 64;

  v8f acc[4];
  #pragma unroll
  for (int j = 0; j < 4; ++j) acc[j] = zero8();

  for (int k0 = 0; k0 < K; k0 += 32) {
    __syncthreads();
    // Stage A: 64 rows x 32 k of bf16 = 256 chunks of 8 elems (16B each).
    #pragma unroll
    for (int c = tid; c < 256; c += 128) {
      int rr = c >> 2, kc = (c & 3) * 8;
      int gm = m0 + rr;
      uint4 val;
      if (gm < M) val = *(const uint4*)(A + (size_t)gm * K + k0 + kc);
      else        val = make_uint4(0u, 0u, 0u, 0u);
      *(uint4*)(As + rr * 32 + kc) = val;
    }
    // Stage B transposed: 32 k x 64 n f32, read float4 along n, scatter bf16.
    #pragma unroll
    for (int q = tid; q < 512; q += 128) {
      int k = q >> 4, n = (q & 15) * 4;
      int gn = n0 + n;
      const float* bp = B + (size_t)(k0 + k) * Nc + gn;
      float4 f;
      if (gn + 3 < Nc) {
        f = *(const float4*)bp;
      } else {
        f.x = (gn + 0 < Nc) ? bp[0] : 0.f;
        f.y = (gn + 1 < Nc) ? bp[1] : 0.f;
        f.z = (gn + 2 < Nc) ? bp[2] : 0.f;
        f.w = (gn + 3 < Nc) ? bp[3] : 0.f;
      }
      Bs[(n + 0) * 32 + k] = (__bf16)f.x;
      Bs[(n + 1) * 32 + k] = (__bf16)f.y;
      Bs[(n + 2) * 32 + k] = (__bf16)f.z;
      Bs[(n + 3) * 32 + k] = (__bf16)f.w;
    }
    // Prefetch next K-chunk of the weight tile into cache while we compute.
    if (k0 + 32 < K)
      __builtin_prefetch(B + (size_t)(k0 + 32 + (tid & 31)) * Nc + n0, 0, 1);
    __syncthreads();
    v16bf a = load_frag_A(As + (wave * 16 + r) * 32, hh);
    #pragma unroll
    for (int j = 0; j < 4; ++j) {
      v16bf bb = load_frag_B(Bs + (j * 16 + r) * 32, hh);
      acc[j] = wmma_bf16(a, bb, acc[j]);
    }
  }

  #pragma unroll
  for (int j = 0; j < 4; ++j) {
    int gn = n0 + j * 16 + r;
    if (gn >= Nc) continue;
    #pragma unroll
    for (int p = 0; p < 8; ++p) {
      int gm = m0 + wave * 16 + p + 8 * hh;
      if (gm >= M) continue;
      float v = acc[j][p];
      size_t o = (size_t)gm * Nc + gn;
      switch (mode) {
        case EP_F32:            outF[o] = v; break;
        case EP_BF16:           outB[o] = (__bf16)v; break;
        case EP_SILU_BF16:      outB[o] = (__bf16)silu_f(v); break;
        case EP_SIG_F32:        outF[o] = sigm_f(v); break;
        case EP_BIAS_SILU_BF16: outB[o] = (__bf16)silu_f(v + bias[gn]); break;
        case EP_BIAS_ADD:       resid[o] += v + bias[gn]; break;
        case EP_ADD:            resid[o] += v; break;
        case EP_GATE_ADD:       resid[o] += v * gate[o]; break;
        case EP_GATE_ADD_T: {
          int ii = gm / transDim, qq = gm % transDim;
          resid[((size_t)qq * transDim + ii) * Nc + gn] += v * gate[o];
        } break;
      }
    }
  }
}

// ------------------------- triangle batched GEMM ----------------------------
// TRI[i,j,d] = sum_k Abuf[d][i][k] * Bbuf[d][j][k]   (128 batches, 160^3 each)
// Abuf is A-layout [row][k]; Bbuf is Bt-layout [col][k] -> fragments straight
// from global with 128-bit loads, no LDS. One wave per 16x16 output tile.

__global__ __launch_bounds__(32) void tri_gemm_k(
    const __bf16* __restrict__ Abuf, const __bf16* __restrict__ Bbuf,
    float* __restrict__ TRI)
{
  const int bj = blockIdx.x, bi = blockIdx.y, d = blockIdx.z;
  const int lane = threadIdx.x, r = lane & 15, hh = lane >> 4;
  const __bf16* Ad = Abuf + (size_t)d * NTOK * NTOK;
  const __bf16* Bd = Bbuf + (size_t)d * NTOK * NTOK;
  v8f acc = zero8();
  for (int kk = 0; kk < NTOK; kk += 32) {
    v16bf a = load_frag_A(Ad + (size_t)(bi * 16 + r) * NTOK + kk, hh);
    v16bf b = load_frag_B(Bd + (size_t)(bj * 16 + r) * NTOK + kk, hh);
    acc = wmma_bf16(a, b, acc);
  }
  const int j = bj * 16 + r;
  #pragma unroll
  for (int p = 0; p < 8; ++p) {
    int i = bi * 16 + p + 8 * hh;
    TRI[((size_t)i * NTOK + j) * DP_ + d] = acc[p];
  }
}

// ------------------------------ LayerNorm -----------------------------------

__global__ __launch_bounds__(128) void ln_rows_k(
    const float* __restrict__ x, const float* __restrict__ g,
    const float* __restrict__ b, __bf16* __restrict__ out, int C)
{
  __shared__ float red[128];
  const int row = blockIdx.x, tid = threadIdx.x;
  const float* xr = x + (size_t)row * C;
  float s = 0.f, s2 = 0.f;
  for (int c = tid; c < C; c += 128) { float v = xr[c]; s += v; s2 += v * v; }
  red[tid] = s; __syncthreads();
  for (int o = 64; o; o >>= 1) { if (tid < o) red[tid] += red[tid + o]; __syncthreads(); }
  float mean = red[0] * (1.0f / C); __syncthreads();
  red[tid] = s2; __syncthreads();
  for (int o = 64; o; o >>= 1) { if (tid < o) red[tid] += red[tid + o]; __syncthreads(); }
  float var = red[0] * (1.0f / C) - mean * mean;
  float rstd = rsqrtf(var + 1e-5f);
  __bf16* orow = out + (size_t)row * C;
  for (int c = tid; c < C; c += 128)
    orow[c] = (__bf16)((xr[c] - mean) * rstd * g[c] + b[c]);
}

// --------------------------- small utility kernels --------------------------

__global__ void f32_to_bf16_k(const float* __restrict__ in,
                              __bf16* __restrict__ out, size_t n) {
  size_t i = (size_t)blockIdx.x * blockDim.x + threadIdx.x;
  if (i < n) out[i] = (__bf16)in[i];
}

__global__ __launch_bounds__(128) void transpose_pair_k(
    const float* __restrict__ in, float* __restrict__ out) {
  int rr = blockIdx.x; int i = rr / NTOK, j = rr % NTOK;
  out[((size_t)j * NTOK + i) * DP_ + threadIdx.x] = in[(size_t)rr * DP_ + threadIdx.x];
}

// AB[25600][256] (silu'd a|b) -> Abuf[d][i][k], Bbuf[d][j][k]
__global__ __launch_bounds__(128) void repack_tri_k(
    const __bf16* __restrict__ AB, __bf16* __restrict__ Abuf,
    __bf16* __restrict__ Bbuf, int outgoing)
{
  int rr = blockIdx.x; int first = rr / NTOK, second = rr % NTOK;
  int idx1, k;
  if (outgoing) { idx1 = first; k = second; }   // AB row = i*160+k (and j*160+k)
  else          { k = first; idx1 = second; }   // AB row = k*160+i (and k*160+j)
  int d = threadIdx.x;
  Abuf[((size_t)d * NTOK + idx1) * NTOK + k] = AB[(size_t)rr * 256 + d];
  Bbuf[((size_t)d * NTOK + idx1) * NTOK + k] = AB[(size_t)rr * 256 + 128 + d];
}

// ------------------------ triangle attention kernel -------------------------
// One block per (row i, head). 320 threads = 10 waves, one 16-q tile per wave.
// scores: WMMA with d_h=16 zero-padded to K=32; softmax over k via half-wave
// shuffles (C-layout: lane=col); LDS round-trip to A-layout; W@V via WMMA.

__global__ __launch_bounds__(320) void attn_pair_k(
    const __bf16* __restrict__ Q, const __bf16* __restrict__ Kt_g,
    const __bf16* __restrict__ V, const float* __restrict__ BIAS,
    __bf16* __restrict__ AO, float scale)
{
  __shared__ __align__(16) __bf16 smem[28160];  // 56,320 B
  __bf16* Vt = smem;            // [16][160]  (persistent)
  __bf16* Kt = smem + 2560;     // [160][32]  (phase 1)
  __bf16* SC = smem + 2560;     // [160][160] (phase 2, overlaps Kt)
  const int i = blockIdx.x, head = blockIdx.y;
  const int tid = threadIdx.x, wave = tid >> 5, lane = tid & 31;
  const int r = lane & 15, hh = lane >> 4;
  const size_t row0 = (size_t)i * NTOK;

  for (int idx = tid; idx < NTOK * 32; idx += 320) {
    int k = idx >> 5, c = idx & 31;
    Kt[idx] = (c < 16) ? Kt_g[(row0 + k) * DP_ + head * 16 + c] : (__bf16)0.0f;
  }
  for (int idx = tid; idx < 16 * NTOK; idx += 320) {
    int dd = idx / NTOK, k = idx % NTOK;
    Vt[idx] = V[(row0 + k) * DP_ + head * 16 + dd];
  }
  __syncthreads();

  const int qbase = wave * 16;
  union { v16bf v; uint4 u[2]; } af;
  af.u[0] = *(const uint4*)(Q + (row0 + qbase + r) * DP_ + head * 16 + 8 * hh);
  af.u[1] = make_uint4(0u, 0u, 0u, 0u);  // K=16..31 pad

  v8f acc[10];
  #pragma unroll
  for (int t = 0; t < 10; ++t) {
    v16bf bb = load_frag_B(Kt + (size_t)(t * 16 + r) * 32, hh);
    acc[t] = wmma_bf16(af.v, bb, zero8());
  }

  float rmax[8], rsum[8];
  #pragma unroll
  for (int p = 0; p < 8; ++p) rmax[p] = -1e30f;
  for (int t = 0; t < 10; ++t) {
    #pragma unroll
    for (int p = 0; p < 8; ++p) {
      int q = qbase + p + 8 * hh, k = t * 16 + r;
      float v = acc[t][p] * scale + BIAS[((size_t)q * NTOK + k) * NHEAD + head];
      acc[t][p] = v;
      rmax[p] = fmaxf(rmax[p], v);
    }
  }
  for (int off = 1; off < 16; off <<= 1) {
    #pragma unroll
    for (int p = 0; p < 8; ++p) rmax[p] = fmaxf(rmax[p], __shfl_xor(rmax[p], off, 32));
  }
  #pragma unroll
  for (int p = 0; p < 8; ++p) rsum[p] = 0.f;
  for (int t = 0; t < 10; ++t) {
    #pragma unroll
    for (int p = 0; p < 8; ++p) {
      float e = __expf(acc[t][p] - rmax[p]); acc[t][p] = e; rsum[p] += e;
    }
  }
  for (int off = 1; off < 16; off <<= 1) {
    #pragma unroll
    for (int p = 0; p < 8; ++p) rsum[p] += __shfl_xor(rsum[p], off, 32);
  }
  #pragma unroll
  for (int p = 0; p < 8; ++p) rsum[p] = 1.0f / rsum[p];

  __syncthreads();  // all waves done with Kt before SC overwrites it
  for (int t = 0; t < 10; ++t) {
    #pragma unroll
    for (int p = 0; p < 8; ++p)
      SC[(size_t)(qbase + p + 8 * hh) * NTOK + t * 16 + r] =
          (__bf16)(acc[t][p] * rsum[p]);
  }
  __syncthreads();

  v8f o = zero8();
  for (int kk = 0; kk < NTOK; kk += 32) {
    v16bf a  = load_frag_A(SC + (size_t)(qbase + r) * NTOK + kk, hh);
    v16bf bb = load_frag_B(Vt + (size_t)r * NTOK + kk, hh);
    o = wmma_bf16(a, bb, o);
  }
  #pragma unroll
  for (int p = 0; p < 8; ++p)
    AO[(row0 + qbase + p + 8 * hh) * DP_ + head * 16 + r] = (__bf16)o[p];
}

// --------------------- single (pair-biased) attention -----------------------
// d_h = 48: 2 K-chunks for scores (zero-padded to 64), 3 output col tiles.
// One block per (head, q-half); 160 threads = 5 waves.

__global__ __launch_bounds__(160) void attn_single_k(
    const __bf16* __restrict__ Q, const __bf16* __restrict__ Kt_g,
    const __bf16* __restrict__ V, const float* __restrict__ BIAS,
    __bf16* __restrict__ AO, float scale)
{
  __shared__ __align__(16) __bf16 smem[20480];  // 40,960 B
  __bf16* Vt = smem;            // [48][160]  (persistent)
  __bf16* Kt = smem + 7680;     // [160][64]  (phase 1)
  __bf16* SC = smem + 7680;     // [80][160]  (phase 2)
  const int head = blockIdx.x, qh = blockIdx.y;
  const int tid = threadIdx.x, wave = tid >> 5, lane = tid & 31;
  const int r = lane & 15, hh = lane >> 4;

  for (int idx = tid; idx < NTOK * 64; idx += 160) {
    int k = idx >> 6, c = idx & 63;
    Kt[idx] = (c < 48) ? Kt_g[(size_t)k * DS_ + head * 48 + c] : (__bf16)0.0f;
  }
  for (int idx = tid; idx < 48 * NTOK; idx += 160) {
    int dd = idx / NTOK, k = idx % NTOK;
    Vt[idx] = V[(size_t)k * DS_ + head * 48 + dd];
  }
  __syncthreads();

  const int qbase = qh * 80 + wave * 16;
  const int q_r = qbase + r;
  union { v16bf v; uint4 u[2]; } af0, af1;
  const __bf16* qp = Q + (size_t)q_r * DS_ + head * 48;
  af0.u[0] = *(const uint4*)(qp + 8 * hh);
  af0.u[1] = *(const uint4*)(qp + 16 + 8 * hh);
  af1.u[0] = *(const uint4*)(qp + 32 + 8 * hh);   // channels 32..47 (valid)
  af1.u[1] = make_uint4(0u, 0u, 0u, 0u);          // channels 48..63 pad

  v8f acc[10];
  #pragma unroll
  for (int t = 0; t < 10; ++t) {
    v8f z = zero8();
    v16bf b0 = load_frag_B(Kt + (size_t)(t * 16 + r) * 64, hh);
    z = wmma_bf16(af0.v, b0, z);
    v16bf b1 = load_frag_B(Kt + (size_t)(t * 16 + r) * 64 + 32, hh);
    acc[t] = wmma_bf16(af1.v, b1, z);
  }

  float rmax[8], rsum[8];
  #pragma unroll
  for (int p = 0; p < 8; ++p) rmax[p] = -1e30f;
  for (int t = 0; t < 10; ++t) {
    #pragma unroll
    for (int p = 0; p < 8; ++p) {
      int q = qbase + p + 8 * hh, k = t * 16 + r;
      float v = acc[t][p] * scale + BIAS[((size_t)q * NTOK + k) * NHEAD + head];
      acc[t][p] = v;
      rmax[p] = fmaxf(rmax[p], v);
    }
  }
  for (int off = 1; off < 16; off <<= 1) {
    #pragma unroll
    for (int p = 0; p < 8; ++p) rmax[p] = fmaxf(rmax[p], __shfl_xor(rmax[p], off, 32));
  }
  #pragma unroll
  for (int p = 0; p < 8; ++p) rsum[p] = 0.f;
  for (int t = 0; t < 10; ++t) {
    #pragma unroll
    for (int p = 0; p < 8; ++p) {
      float e = __expf(acc[t][p] - rmax[p]); acc[t][p] = e; rsum[p] += e;
    }
  }
  for (int off = 1; off < 16; off <<= 1) {
    #pragma unroll
    for (int p = 0; p < 8; ++p) rsum[p] += __shfl_xor(rsum[p], off, 32);
  }
  #pragma unroll
  for (int p = 0; p < 8; ++p) rsum[p] = 1.0f / rsum[p];

  __syncthreads();
  for (int t = 0; t < 10; ++t) {
    #pragma unroll
    for (int p = 0; p < 8; ++p)
      SC[(size_t)(wave * 16 + p + 8 * hh) * NTOK + t * 16 + r] =
          (__bf16)(acc[t][p] * rsum[p]);
  }
  __syncthreads();

  for (int ct = 0; ct < 3; ++ct) {
    v8f o = zero8();
    for (int kk = 0; kk < NTOK; kk += 32) {
      v16bf a  = load_frag_A(SC + (size_t)(wave * 16 + r) * NTOK + kk, hh);
      v16bf bb = load_frag_B(Vt + (size_t)(ct * 16 + r) * NTOK + kk, hh);
      o = wmma_bf16(a, bb, o);
    }
    #pragma unroll
    for (int p = 0; p < 8; ++p)
      AO[(size_t)(qbase + p + 8 * hh) * DS_ + head * 48 + ct * 16 + r] = (__bf16)o[p];
  }
}

// ------------------------------ host helpers --------------------------------

static void gemm(hipStream_t st, const __bf16* A, const float* B,
                 int M, int K, int Nc, int mode,
                 float* outF = nullptr, __bf16* outB = nullptr,
                 const float* bias = nullptr, const float* gate = nullptr,
                 float* resid = nullptr, int transDim = NTOK)
{
  dim3 g((Nc + 63) / 64, (M + 63) / 64);
  gemm_bf16_k<<<g, 128, 0, st>>>(A, B, M, K, Nc, outF, outB, bias, gate,
                                 resid, mode, transDim);
}

static void tri_mult_step(hipStream_t st, float* P,
    const float* lns, const float* lnb, const float* Win,
    const float* Wg, const float* Wo,
    __bf16* XBF, __bf16* ABBF, float* GF, float* TRIF,
    __bf16* ABUF, __bf16* BBUF, int outgoing)
{
  ln_rows_k<<<NN_, 128, 0, st>>>(P, lns, lnb, XBF, DP_);
  gemm(st, XBF, Win, NN_, DP_, 2 * DP_, EP_SILU_BF16, nullptr, ABBF);
  gemm(st, XBF, Wg,  NN_, DP_, DP_, EP_SIG_F32, GF);
  repack_tri_k<<<NN_, 128, 0, st>>>(ABBF, ABUF, BBUF, outgoing);
  dim3 tg(10, 10, DP_);
  tri_gemm_k<<<tg, 32, 0, st>>>(ABUF, BBUF, TRIF);
  ln_rows_k<<<NN_, 128, 0, st>>>(TRIF, lns, lnb, XBF, DP_);   // same LN params
  gemm(st, XBF, Wo, NN_, DP_, DP_, EP_GATE_ADD, nullptr, nullptr, nullptr, GF, P);
}

static void tri_att_step(hipStream_t st, float* P, float* PT,
    const float* Wq, const float* Wk, const float* Wv, const float* Wo,
    const float* Wg, const float* Wb,
    __bf16* XBF, __bf16* QB, __bf16* KB, __bf16* VB,
    float* GF, float* BIASF, __bf16* AOBF, int ending)
{
  const float* Psrc = P;
  if (ending) { transpose_pair_k<<<NN_, 128, 0, st>>>(P, PT); Psrc = PT; }
  f32_to_bf16_k<<<(unsigned)((PAIR_ELEMS + 255) / 256), 256, 0, st>>>(Psrc, XBF, PAIR_ELEMS);
  gemm(st, XBF, Wq, NN_, DP_, DP_, EP_BF16, nullptr, QB);
  gemm(st, XBF, Wk, NN_, DP_, DP_, EP_BF16, nullptr, KB);
  gemm(st, XBF, Wv, NN_, DP_, DP_, EP_BF16, nullptr, VB);
  gemm(st, XBF, Wb, NN_, DP_, NHEAD, EP_F32, BIASF);
  gemm(st, XBF, Wg, NN_, DP_, DP_, EP_SIG_F32, GF);
  dim3 ag(NTOK, NHEAD);
  attn_pair_k<<<ag, 320, 0, st>>>(QB, KB, VB, BIASF, AOBF,
                                  0.08838834764831845f /* DP^-0.5 */);
  if (!ending)
    gemm(st, AOBF, Wo, NN_, DP_, DP_, EP_GATE_ADD, nullptr, nullptr, nullptr, GF, P);
  else
    gemm(st, AOBF, Wo, NN_, DP_, DP_, EP_GATE_ADD_T, nullptr, nullptr, nullptr, GF, P, NTOK);
}

static void apb_step(hipStream_t st, float* S, const float* P,
    const float* Wq, const float* Wk, const float* Wv, const float* Wo,
    const float* Wb, const float* lns, const float* lnb,
    __bf16* SBF, __bf16* QS, __bf16* KS, __bf16* VS, __bf16* AOS,
    __bf16* XBF, float* BIASF)
{
  f32_to_bf16_k<<<(unsigned)((SING_ELEMS + 255) / 256), 256, 0, st>>>(S, SBF, SING_ELEMS);
  gemm(st, SBF, Wq, NTOK, DS_, DS_, EP_BF16, nullptr, QS);
  gemm(st, SBF, Wk, NTOK, DS_, DS_, EP_BF16, nullptr, KS);
  gemm(st, SBF, Wv, NTOK, DS_, DS_, EP_BF16, nullptr, VS);
  ln_rows_k<<<NN_, 128, 0, st>>>(P, lns, lnb, XBF, DP_);
  gemm(st, XBF, Wb, NN_, DP_, NHEAD, EP_F32, BIASF);
  dim3 ag(NHEAD, 2);
  attn_single_k<<<ag, 160, 0, st>>>(QS, KS, VS, BIASF, AOS,
                                    0.05103103630798287f /* DS^-0.5 */);
  gemm(st, AOS, Wo, NTOK, DS_, DS_, EP_ADD, nullptr, nullptr, nullptr, nullptr, S);
}

static void transition_step(hipStream_t st, float* S,
    const float* lns, const float* lnb, const float* W1, const float* b1,
    const float* W2, const float* b2, __bf16* SBF, __bf16* H1)
{
  ln_rows_k<<<NTOK, 128, 0, st>>>(S, lns, lnb, SBF, DS_);
  gemm(st, SBF, W1, NTOK, DS_, 4 * DS_, EP_BIAS_SILU_BF16, nullptr, H1, b1);
  gemm(st, H1, W2, NTOK, 4 * DS_, DS_, EP_BIAS_ADD, nullptr, nullptr, b2, nullptr, S);
}

// ------------------------------- entry point --------------------------------
// Input order: single_repr, pair_repr, then params pytree leaves in JAX
// (alphabetical dict-key) order:
//   apb:     Wb Wk Wo Wq Wv ln_b ln_s
//   ta_end:  Wb Wg Wk Wo Wq Wv
//   ta_start:Wb Wg Wk Wo Wq Wv
//   tm_in:   W_g W_in W_o ln_b ln_s
//   tm_out:  W_g W_in W_o ln_b ln_s
//   trans:   W1 W2 b1 b2 ln_b ln_s

extern "C" void kernel_launch(void* const* d_in, const int* in_sizes, int n_in,
                              void* d_out, int out_size, void* d_ws, size_t ws_size,
                              hipStream_t stream)
{
  (void)in_sizes; (void)n_in; (void)out_size; (void)ws_size;
  const float* in_single = (const float*)d_in[0];
  const float* in_pair   = (const float*)d_in[1];
  const float* apb_Wb  = (const float*)d_in[2];
  const float* apb_Wk  = (const float*)d_in[3];
  const float* apb_Wo  = (const float*)d_in[4];
  const float* apb_Wq  = (const float*)d_in[5];
  const float* apb_Wv  = (const float*)d_in[6];
  const float* apb_lnb = (const float*)d_in[7];
  const float* apb_lns = (const float*)d_in[8];
  const float* tae_Wb  = (const float*)d_in[9];
  const float* tae_Wg  = (const float*)d_in[10];
  const float* tae_Wk  = (const float*)d_in[11];
  const float* tae_Wo  = (const float*)d_in[12];
  const float* tae_Wq  = (const float*)d_in[13];
  const float* tae_Wv  = (const float*)d_in[14];
  const float* tas_Wb  = (const float*)d_in[15];
  const float* tas_Wg  = (const float*)d_in[16];
  const float* tas_Wk  = (const float*)d_in[17];
  const float* tas_Wo  = (const float*)d_in[18];
  const float* tas_Wq  = (const float*)d_in[19];
  const float* tas_Wv  = (const float*)d_in[20];
  const float* tmi_Wg  = (const float*)d_in[21];
  const float* tmi_Win = (const float*)d_in[22];
  const float* tmi_Wo  = (const float*)d_in[23];
  const float* tmi_lnb = (const float*)d_in[24];
  const float* tmi_lns = (const float*)d_in[25];
  const float* tmo_Wg  = (const float*)d_in[26];
  const float* tmo_Win = (const float*)d_in[27];
  const float* tmo_Wo  = (const float*)d_in[28];
  const float* tmo_lnb = (const float*)d_in[29];
  const float* tmo_lns = (const float*)d_in[30];
  const float* tr_W1   = (const float*)d_in[31];
  const float* tr_W2   = (const float*)d_in[32];
  const float* tr_b1   = (const float*)d_in[33];
  const float* tr_b2   = (const float*)d_in[34];
  const float* tr_lnb  = (const float*)d_in[35];
  const float* tr_lns  = (const float*)d_in[36];

  char* ws = (char*)d_ws; size_t off = 0;
  auto carve = [&](size_t bytes) -> char* {
    char* p = ws + off; off += (bytes + 255) & ~(size_t)255; return p;
  };
  float*  P     = (float*)carve(PAIR_ELEMS * 4);
  float*  S     = (float*)carve(SING_ELEMS * 4);
  float*  PT    = (float*)carve(PAIR_ELEMS * 4);
  float*  GF    = (float*)carve(PAIR_ELEMS * 4);
  float*  TRIF  = (float*)carve(PAIR_ELEMS * 4);
  float*  BIASF = (float*)carve((size_t)NN_ * NHEAD * 4);
  __bf16* XBF   = (__bf16*)carve((size_t)NN_ * 256 * 2);
  __bf16* ABBF  = (__bf16*)carve((size_t)NN_ * 256 * 2);   // also attention-out
  __bf16* QB    = (__bf16*)carve(PAIR_ELEMS * 2);
  __bf16* KB    = (__bf16*)carve(PAIR_ELEMS * 2);
  __bf16* VB    = (__bf16*)carve(PAIR_ELEMS * 2);
  __bf16* ABUF  = (__bf16*)carve(PAIR_ELEMS * 2);
  __bf16* BBUF  = (__bf16*)carve(PAIR_ELEMS * 2);
  __bf16* SBF   = (__bf16*)carve(SING_ELEMS * 2);
  __bf16* QS    = (__bf16*)carve(SING_ELEMS * 2);
  __bf16* KS    = (__bf16*)carve(SING_ELEMS * 2);
  __bf16* VS    = (__bf16*)carve(SING_ELEMS * 2);
  __bf16* AOS   = (__bf16*)carve(SING_ELEMS * 2);
  __bf16* H1    = (__bf16*)carve((size_t)NTOK * 4 * DS_ * 2);

  (void)hipMemcpyAsync(P, in_pair, PAIR_ELEMS * 4, hipMemcpyDeviceToDevice, stream);
  (void)hipMemcpyAsync(S, in_single, SING_ELEMS * 4, hipMemcpyDeviceToDevice, stream);

  for (int b = 0; b < 2; ++b) {
    const size_t oPP = (size_t)b * DP_ * DP_;     // 128x128 weights
    const size_t oPI = (size_t)b * DP_ * 2 * DP_; // 128x256
    const size_t oPB = (size_t)b * DP_ * NHEAD;   // 128x8
    const size_t oP  = (size_t)b * DP_;           // 128 ln vecs
    const size_t oSS = (size_t)b * DS_ * DS_;     // 384x384
    const size_t oS  = (size_t)b * DS_;
    const size_t oW1 = (size_t)b * DS_ * 4 * DS_;
    const size_t oB1 = (size_t)b * 4 * DS_;

    tri_mult_step(stream, P, tmo_lns + oP, tmo_lnb + oP, tmo_Win + oPI,
                  tmo_Wg + oPP, tmo_Wo + oPP, XBF, ABBF, GF, TRIF, ABUF, BBUF, 1);
    tri_mult_step(stream, P, tmi_lns + oP, tmi_lnb + oP, tmi_Win + oPI,
                  tmi_Wg + oPP, tmi_Wo + oPP, XBF, ABBF, GF, TRIF, ABUF, BBUF, 0);
    tri_att_step(stream, P, PT, tas_Wq + oPP, tas_Wk + oPP, tas_Wv + oPP,
                 tas_Wo + oPP, tas_Wg + oPP, tas_Wb + oPB,
                 XBF, QB, KB, VB, GF, BIASF, ABBF, 0);
    tri_att_step(stream, P, PT, tae_Wq + oPP, tae_Wk + oPP, tae_Wv + oPP,
                 tae_Wo + oPP, tae_Wg + oPP, tae_Wb + oPB,
                 XBF, QB, KB, VB, GF, BIASF, ABBF, 1);
    apb_step(stream, S, P, apb_Wq + oSS, apb_Wk + oSS, apb_Wv + oSS,
             apb_Wo + oSS, apb_Wb + oPB, apb_lns + oP, apb_lnb + oP,
             SBF, QS, KS, VS, AOS, XBF, BIASF);
    transition_step(stream, S, tr_lns + oS, tr_lnb + oS, tr_W1 + oW1,
                    tr_b1 + oB1, tr_W2 + oW1, tr_b2 + oS, SBF, H1);
  }

  (void)hipMemcpyAsync((float*)d_out, S, SING_ELEMS * 4,
                       hipMemcpyDeviceToDevice, stream);
  (void)hipMemcpyAsync((float*)d_out + SING_ELEMS, P, PAIR_ELEMS * 4,
                       hipMemcpyDeviceToDevice, stream);
}
